// FixedPredLayer_64158221468274
// MI455X (gfx1250) — matrix operand order repaired and checked by reading, compiled, not run on hardware
//
#include <hip/hip_runtime.h>
#include <math.h>

// ---------------------------------------------------------------------------
// Problem constants
// ---------------------------------------------------------------------------
#define NROWS   4096        // M
#define DIM     1024        // K
#define VWORDS  50257       // V (true vocab)
#define VPAD    50304       // V padded to 128*393
#define BM      128         // block tile M
#define BN      128         // block tile V
#define KB      32          // K step = one bf16 WMMA depth
#define ITERS   (DIM / KB)  // 32
#define LDSTR   40          // LDS row stride in bf16 (80B, spreads banks)

typedef __attribute__((ext_vector_type(16))) __bf16 bf16x16;
typedef __attribute__((ext_vector_type(8)))  float  f32x8;
typedef __attribute__((ext_vector_type(4)))  unsigned int u32x4;
typedef __attribute__((ext_vector_type(8)))  int    i32x8;
typedef __attribute__((ext_vector_type(4)))  int    i32x4;

// ---------------------------------------------------------------------------
// fp32 -> bf16 (round to nearest even)
// ---------------------------------------------------------------------------
__device__ inline unsigned short f32_to_bf16(float f) {
    union { float f; unsigned u; } c; c.f = f;
    unsigned u = c.u;
    u += 0x7FFFu + ((u >> 16) & 1u);
    return (unsigned short)(u >> 16);
}

// ---------------------------------------------------------------------------
// Convert x [4096 x 1024] fp32 -> bf16
// ---------------------------------------------------------------------------
__global__ __launch_bounds__(256) void cvt_x_kernel(const float* __restrict__ x,
                                                    unsigned short* __restrict__ xb) {
    size_t i = ((size_t)blockIdx.x * blockDim.x + threadIdx.x) * 4;
    if (i >= (size_t)NROWS * DIM) return;
    const float4 v = *reinterpret_cast<const float4*>(x + i);
    ushort4 o;
    o.x = f32_to_bf16(v.x); o.y = f32_to_bf16(v.y);
    o.z = f32_to_bf16(v.z); o.w = f32_to_bf16(v.w);
    *reinterpret_cast<ushort4*>(xb + i) = o;
}

// ---------------------------------------------------------------------------
// Convert W fp32 -> bf16, padded to [50304 x 1024], pad rows zeroed
// ---------------------------------------------------------------------------
__global__ __launch_bounds__(256) void cvt_w_kernel(const float* __restrict__ W,
                                                    unsigned short* __restrict__ Wb) {
    size_t i = ((size_t)blockIdx.x * blockDim.x + threadIdx.x) * 4;
    if (i >= (size_t)VPAD * DIM) return;
    size_t row = i >> 10;
    ushort4 o;
    if (row < (size_t)VWORDS) {
        const float4 v = *reinterpret_cast<const float4*>(W + i);
        o.x = f32_to_bf16(v.x); o.y = f32_to_bf16(v.y);
        o.z = f32_to_bf16(v.z); o.w = f32_to_bf16(v.w);
    } else {
        o.x = 0; o.y = 0; o.z = 0; o.w = 0;
    }
    *reinterpret_cast<ushort4*>(Wb + i) = o;
}

// ---------------------------------------------------------------------------
// TDM descriptor build + issue: 2D tile [BN rows x KB cols] of bf16 from a
// row-major tensor (row stride DIM) into LDS, with a 4-DWORD pad every
// 16 DWORDs (=> LDS row stride 40 bf16, matching Bs[.][LDSTR]).
// clang-23 / therock-10.0 6-arg builtin form.
// ---------------------------------------------------------------------------
__device__ inline void tdm_load_tile_b(const __bf16* gsrc, unsigned lds_byte_addr) {
    unsigned long long ga = (unsigned long long)(const void*)gsrc;
    u32x4 g0;
    g0.x = 1u;                                            // count=1, user D#
    g0.y = lds_byte_addr;                                 // lds_addr
    g0.z = (unsigned)(ga & 0xFFFFFFFFu);                  // global_addr[31:0]
    g0.w = (unsigned)((ga >> 32) & 0x01FFFFFFu)           // global_addr[56:32]
         | 0x80000000u;                                   // type = 2 (image)
    i32x8 g1;
    g1[0] = (1 << 16)        // data_size = 1 (2 bytes)
          | (1 << 20)        // pad_enable
          | (3 << 22)        // pad_interval: 16 DWORDs
          | (3 << 25);       // pad_amount: 4 DWORDs
    g1[1] = (int)((DIM & 0xFFFF) << 16);                  // tensor_dim0 lo16
    g1[2] = (int)((DIM >> 16) | ((VPAD & 0xFFFF) << 16)); // dim0 hi | dim1 lo
    g1[3] = (int)((VPAD >> 16) | (KB << 16));             // dim1 hi | tile_dim0
    g1[4] = BN;                                           // tile_dim1 rows
    g1[5] = DIM;                                          // tensor_dim0_stride lo32
    g1[6] = 0;
    g1[7] = 0;
    i32x4 z4 = {0, 0, 0, 0};                              // groups 2/3 unused (2D)
    i32x8 z8 = {0, 0, 0, 0, 0, 0, 0, 0};                  // extra group (unused)
    __builtin_amdgcn_tensor_load_to_lds(g0, g1, z4, z4, z8, 0);
}

// ---------------------------------------------------------------------------
// scores = x @ W^T.  Block 256 threads (8 wave32), tile 128x128.
// Waves: 4(M) x 2(N); wave tile 32x64 = 2x4 WMMA accumulators.
// A staged via global_load->ds_store (single buffer), B via TDM (double buf).
// ---------------------------------------------------------------------------
__global__ __launch_bounds__(256) void wmma_gemm_kernel(const __bf16* __restrict__ Xb,
                                                        const __bf16* __restrict__ Wb,
                                                        float* __restrict__ scores) {
    __shared__ __align__(16) __bf16 As[BM][LDSTR];
    __shared__ __align__(16) __bf16 Bs[2][BN][LDSTR];

    const int tid  = threadIdx.x;
    const int lane = tid & 31;
    const int wave = tid >> 5;
    const int wm   = (wave >> 1) * 32;    // 0,32,64,96
    const int wn   = (wave & 1) * 64;     // 0,64
    const int m0   = blockIdx.x * BM;
    const int v0   = blockIdx.y * BN;

    const int half = lane >> 4;
    const int l16  = lane & 15;
    const int kofs = half * 8;

    f32x8 acc[2][4] = {};

    // prologue: wave 0 kicks off TDM for B tile of iteration 0 into buffer 0
    if (wave == 0) {
        tdm_load_tile_b(&Wb[(size_t)v0 * DIM + 0],
                        (unsigned)(unsigned long long)(const void*)&Bs[0][0][0]);
    }

    for (int it = 0; it < ITERS; ++it) {
        const int kb  = it * KB;
        const int buf = it & 1;

        // ---- stage A tile: 128 rows x 32 bf16 = 512 x 16B chunks ----
        #pragma unroll
        for (int i = 0; i < 2; ++i) {
            const int chunk = tid + i * 256;
            const int row   = chunk >> 2;
            const int col   = (chunk & 3) * 8;
            *reinterpret_cast<uint4*>(&As[row][col]) =
                *reinterpret_cast<const uint4*>(&Xb[(size_t)(m0 + row) * DIM + kb + col]);
        }
        if (kb + KB < DIM)
            __builtin_prefetch(&Xb[(size_t)(m0 + (tid >> 2)) * DIM + kb + KB], 0, 1);

        // ---- wave 0: issue TDM for next B tile, then wait for current ----
        if (wave == 0) {
            if (it + 1 < ITERS) {
                tdm_load_tile_b(&Wb[(size_t)v0 * DIM + kb + KB],
                                (unsigned)(unsigned long long)(const void*)&Bs[buf ^ 1][0][0]);
                __builtin_amdgcn_s_wait_tensorcnt(1);   // current tile done
            } else {
                __builtin_amdgcn_s_wait_tensorcnt(0);
            }
        }
        __syncthreads();   // A stores visible, Bs[buf] DMA complete

        // ---- fragments ----
        bf16x16 afrag[2], bfrag[4];
        #pragma unroll
        for (int mi = 0; mi < 2; ++mi) {
            const int m = wm + mi * 16 + l16;
            union { bf16x16 v; uint4 u[2]; } t;
            t.u[0] = *reinterpret_cast<const uint4*>(&As[m][kofs]);
            t.u[1] = *reinterpret_cast<const uint4*>(&As[m][16 + kofs]);
            afrag[mi] = t.v;
        }
        #pragma unroll
        for (int ni = 0; ni < 4; ++ni) {
            const int v = wn + ni * 16 + l16;
            union { bf16x16 v; uint4 u[2]; } t;
            t.u[0] = *reinterpret_cast<const uint4*>(&Bs[buf][v][half * 8]);
            t.u[1] = *reinterpret_cast<const uint4*>(&Bs[buf][v][16 + half * 8]);
            bfrag[ni] = t.v;
        }

        // ---- 8 WMMAs per wave per K-step ----
        #pragma unroll
        for (int mi = 0; mi < 2; ++mi)
            #pragma unroll
            for (int ni = 0; ni < 4; ++ni)
                acc[mi][ni] = __builtin_amdgcn_wmma_f32_16x16x32_bf16(
                    false, afrag[mi], false, bfrag[ni],
                    (short)0, acc[mi][ni], false, false);

        __syncthreads();   // protect As (and Bs parity) before next overwrite
    }

    // ---- store D: VGPR r -> M = r + 8*half, N = l16 ----
    #pragma unroll
    for (int mi = 0; mi < 2; ++mi) {
        #pragma unroll
        for (int ni = 0; ni < 4; ++ni) {
            const int n = v0 + wn + ni * 16 + l16;
            if (n < VWORDS) {
                const int mbase = m0 + wm + mi * 16 + half * 8;
                #pragma unroll
                for (int r = 0; r < 8; ++r)
                    scores[(size_t)(mbase + r) * VWORDS + n] = acc[mi][ni][r];
            }
        }
    }
}

// ---------------------------------------------------------------------------
// Per-row logsumexp + target gather
// ---------------------------------------------------------------------------
__global__ __launch_bounds__(256) void lse_loss_kernel(const float* __restrict__ scores,
                                                       const int* __restrict__ y,
                                                       float* __restrict__ loss) {
    const int row = blockIdx.x;
    const float* s = scores + (size_t)row * VWORDS;
    __shared__ float red[256];

    float m = -__builtin_inff();
    for (int v = threadIdx.x; v < VWORDS; v += 256) m = fmaxf(m, s[v]);
    red[threadIdx.x] = m;
    __syncthreads();
    for (int off = 128; off > 0; off >>= 1) {
        if (threadIdx.x < off)
            red[threadIdx.x] = fmaxf(red[threadIdx.x], red[threadIdx.x + off]);
        __syncthreads();
    }
    m = red[0];
    __syncthreads();

    float sum = 0.0f;
    for (int v = threadIdx.x; v < VWORDS; v += 256) sum += __expf(s[v] - m);
    red[threadIdx.x] = sum;
    __syncthreads();
    for (int off = 128; off > 0; off >>= 1) {
        if (threadIdx.x < off)
            red[threadIdx.x] += red[threadIdx.x + off];
        __syncthreads();
    }

    if (threadIdx.x == 0)
        loss[row] = m + __logf(red[0]) - s[y[row]];
}

// ---------------------------------------------------------------------------
// Host launcher
// ---------------------------------------------------------------------------
extern "C" void kernel_launch(void* const* d_in, const int* in_sizes, int n_in,
                              void* d_out, int out_size, void* d_ws, size_t ws_size,
                              hipStream_t stream) {
    (void)in_sizes; (void)n_in; (void)out_size; (void)ws_size;

    const float* x = (const float*)d_in[0];
    const float* W = (const float*)d_in[1];
    const int*   y = (const int*)d_in[2];

    float* scores = (float*)d_out;
    float* loss   = scores + (size_t)NROWS * VWORDS;

    unsigned short* xb = (unsigned short*)d_ws;
    unsigned short* Wb = xb + (size_t)NROWS * DIM;

    {
        const size_t nx = (size_t)NROWS * DIM / 4;
        cvt_x_kernel<<<(unsigned)((nx + 255) / 256), 256, 0, stream>>>(x, xb);
        const size_t nw = (size_t)VPAD * DIM / 4;
        cvt_w_kernel<<<(unsigned)((nw + 255) / 256), 256, 0, stream>>>(W, Wb);
    }

    {
        dim3 grid(NROWS / BM, VPAD / BN);
        wmma_gemm_kernel<<<grid, 256, 0, stream>>>(
            (const __bf16*)xb, (const __bf16*)Wb, scores);
    }

    lse_loss_kernel<<<NROWS, 256, 0, stream>>>(scores, y, loss);
}
